// PastHead_62620623175864
// MI455X (gfx1250) — compile-verified, matrix operand order checked by reading, not run on hardware
//
#include <hip/hip_runtime.h>
#include <hip/hip_bf16.h>
#include <stdint.h>

#define I_DIM 128
#define H_DIM 256
#define G_DIM 1024   // 4*H
#define B_DIM 1024
#define STEPS 511
#define K_DIM 384    // I + H

typedef __attribute__((ext_vector_type(16))) __bf16 bf16x16;
typedef __attribute__((ext_vector_type(8)))  float  f32x8;

#if defined(__gfx1250__) &&                                        \
    __has_builtin(__builtin_amdgcn_global_load_async_to_lds_b128) && \
    __has_builtin(__builtin_amdgcn_s_wait_asynccnt)
#define USE_ASYNC_LDS 1
#else
#define USE_ASYNC_LDS 0
#endif

#if USE_ASYNC_LDS
typedef int i32x4_t __attribute__((vector_size(16)));
typedef __attribute__((address_space(1))) i32x4_t* as1_i32x4p;
typedef __attribute__((address_space(3))) i32x4_t* as3_i32x4p;
#endif

// 16-byte global->LDS copy: async (direct-to-LDS, ASYNCcnt) on CDNA5,
// sync VGPR round-trip fallback otherwise.
__device__ __forceinline__ void cp16(const __bf16* g, __bf16* l) {
#if USE_ASYNC_LDS
  // flat global addr == AS1 addr; flat LDS addr truncated to 32 bits == AS3 offset
  as1_i32x4p gp = (as1_i32x4p)(unsigned long long)(uintptr_t)g;
  as3_i32x4p lp = (as3_i32x4p)(unsigned int)(uintptr_t)l;
  __builtin_amdgcn_global_load_async_to_lds_b128(gp, lp, 0, 0);
#else
  *(uint4*)l = *(const uint4*)g;
#endif
}

__device__ __forceinline__ void wait_async_le4() {
#if USE_ASYNC_LDS
  __builtin_amdgcn_s_wait_asynccnt(4);
#endif
}
__device__ __forceinline__ void wait_async_0() {
#if USE_ASYNC_LDS
  __builtin_amdgcn_s_wait_asynccnt(0);
#endif
}

// ---------------- prologue: pack W = [W_ih | W_hh] -> bf16, bias = b_ih + b_hh
__global__ __launch_bounds__(256) void pack_weights(
    const float* __restrict__ W_ih, const float* __restrict__ W_hh,
    const float* __restrict__ b_ih, const float* __restrict__ b_hh,
    __bf16* __restrict__ Wc, float* __restrict__ bias)
{
  int idx = blockIdx.x * blockDim.x + threadIdx.x;
  if (idx < G_DIM * K_DIM) {
    int g = idx / K_DIM, k = idx % K_DIM;
    float v = (k < I_DIM) ? W_ih[g * I_DIM + k] : W_hh[g * H_DIM + (k - I_DIM)];
    Wc[idx] = (__bf16)v;
  }
  if (idx < G_DIM) bias[idx] = b_ih[idx] + b_hh[idx];
}

// init: ab[b][0:128] = bf16(x[0][b]), ab[b][128:384] = bf16(hx0), c = cx0
__global__ __launch_bounds__(256) void init_state(
    const float* __restrict__ hx0, const float* __restrict__ cx0,
    const float* __restrict__ x0,
    __bf16* __restrict__ ab, float* __restrict__ c)
{
  int idx = blockIdx.x * blockDim.x + threadIdx.x;   // [0, B*H)
  int b = idx >> 8;
  int j = idx & (H_DIM - 1);
  ab[(size_t)b * K_DIM + I_DIM + j] = (__bf16)hx0[j];
  c[idx] = cx0[j];
  if (idx < B_DIM * I_DIM) {
    int b2 = idx >> 7, k = idx & (I_DIM - 1);
    ab[(size_t)b2 * K_DIM + k] = (__bf16)x0[idx];
  }
}

// ---------------- per-step gates GEMM: gates = [x_t | h] @ Wc^T + bias
// block = 128 threads (4 waves), tile = 64x64, K = 384 in 12 chunks of 32,
// double-buffered LDS fed by async global->LDS copies.
__global__ __launch_bounds__(128) void lstm_gates(
    const __bf16* __restrict__ ab,    // [B][K] bf16 (x_t | h)
    const __bf16* __restrict__ Wc,    // [G][K] bf16
    const float*  __restrict__ bias,  // [G]
    float*        __restrict__ gates) // [B][G] fp32
{
  __shared__ __align__(32) __bf16 Atile[2][64][32];
  __shared__ __align__(32) __bf16 Btile[2][64][32];

  const int tid = threadIdx.x;
  const int cb  = blockIdx.x;   // G block (16)
  const int rb  = blockIdx.y;   // B block (16)
  const int w   = tid >> 5;     // wave 0..3
  const int l   = tid & 31;     // lane
  const int ln  = l & 15;
  const int kh  = (l >> 4) * 16;

  f32x8 acc[4];
#pragma unroll
  for (int n = 0; n < 4; ++n) {
    float bv = bias[cb * 64 + n * 16 + ln];
    f32x8 t = {bv, bv, bv, bv, bv, bv, bv, bv};
    acc[n] = t;
  }

  // issue one K-chunk (64x32 A + 64x32 B) into buffer s: 4 async b128/thread
  auto issue = [&](int kc, int s) {
    const int kbase = kc * 32;
#pragma unroll
    for (int it = 0; it < 2; ++it) {
      int e = (tid + it * 128) * 8;      // element index in 64x32 tile
      int row = e >> 5, k = e & 31;
      cp16(&ab[(size_t)(rb * 64 + row) * K_DIM + kbase + k], &Atile[s][row][k]);
      cp16(&Wc[(size_t)(cb * 64 + row) * K_DIM + kbase + k], &Btile[s][row][k]);
    }
  };

  auto compute = [&](int s) {
    bf16x16 a = *(const bf16x16*)&Atile[s][w * 16 + ln][kh];
    bf16x16 bfr[4];
#pragma unroll
    for (int n = 0; n < 4; ++n)
      bfr[n] = *(const bf16x16*)&Btile[s][n * 16 + ln][kh];
#pragma unroll
    for (int n = 0; n < 4; ++n)
      acc[n] = __builtin_amdgcn_wmma_f32_16x16x32_bf16(
          false, a, false, bfr[n], (short)0, acc[n], false, false);
  };

  issue(0, 0);
  for (int kc = 0; kc < K_DIM / 32 - 1; ++kc) {
    issue(kc + 1, (kc + 1) & 1);
    wait_async_le4();                 // chunk kc landed (async loads complete in order)
    __syncthreads();
    compute(kc & 1);
    __syncthreads();                  // safe to overwrite buffer (kc&1) next iter
  }
  wait_async_0();
  __syncthreads();
  compute((K_DIM / 32 - 1) & 1);

  // store: C/D layout — VGPR r: lanes 0-15 -> M=r, lanes 16-31 -> M=8+r
  const int mbase = (l >> 4) * 8;
#pragma unroll
  for (int n = 0; n < 4; ++n) {
    int col = cb * 64 + n * 16 + ln;
#pragma unroll
    for (int r = 0; r < 8; ++r) {
      int row = rb * 64 + w * 16 + mbase + r;
      gates[(size_t)row * G_DIM + col] = acc[n][r];
    }
  }
}

// ---------------- per-step pointwise LSTM cell update (+ stage next x as bf16)
__global__ __launch_bounds__(256) void lstm_pointwise(
    const float* __restrict__ gates, // [B][G]
    float* __restrict__ c,           // [B][H] (d_out second half)
    float* __restrict__ h,           // [B][H] fp32 (d_out first half)
    __bf16* __restrict__ ab,         // [B][K]: h -> cols 128..383
    const float* __restrict__ xnext) // [B][I]: next step's x -> cols 0..127
{
  int idx = blockIdx.x * blockDim.x + threadIdx.x;  // [0, B*H)
  int b = idx >> 8;
  int j = idx & (H_DIM - 1);
  const float* gb = gates + (size_t)b * G_DIM;
  float ig = gb[j];
  float fg = gb[H_DIM + j];
  float gg = gb[2 * H_DIM + j];
  float og = gb[3 * H_DIM + j];
  ig = 1.f / (1.f + __expf(-ig));
  fg = 1.f / (1.f + __expf(-fg));
  og = 1.f / (1.f + __expf(-og));
  gg = tanhf(gg);
  float cn = fg * c[idx] + ig * gg;
  float hn = og * tanhf(cn);
  c[idx] = cn;
  h[idx] = hn;
  ab[(size_t)b * K_DIM + I_DIM + j] = (__bf16)hn;
  if (idx < B_DIM * I_DIM) {
    int b2 = idx >> 7, k = idx & (I_DIM - 1);
    ab[(size_t)b2 * K_DIM + k] = (__bf16)xnext[idx];
  }
}

extern "C" void kernel_launch(void* const* d_in, const int* in_sizes, int n_in,
                              void* d_out, int out_size, void* d_ws, size_t ws_size,
                              hipStream_t stream) {
  const float* x    = (const float*)d_in[0];
  const float* W_ih = (const float*)d_in[1];
  const float* W_hh = (const float*)d_in[2];
  const float* b_ih = (const float*)d_in[3];
  const float* b_hh = (const float*)d_in[4];
  const float* hx0  = (const float*)d_in[5];
  const float* cx0  = (const float*)d_in[6];

  // workspace layout (bytes): Wc bf16 [1024*384] @0, bias f32 [1024] @786432,
  // ab bf16 [1024*384] @790528, gates f32 [1024*1024] @1576960  (~5.6 MB)
  char* ws = (char*)d_ws;
  __bf16* Wc   = (__bf16*)(ws);
  float*  bias = (float*) (ws + 786432);
  __bf16* ab   = (__bf16*)(ws + 790528);
  float*  gate = (float*) (ws + 1576960);

  float* h_out = (float*)d_out;               // final h
  float* c_out = h_out + B_DIM * H_DIM;       // final c (updated in place)

  pack_weights<<<(G_DIM * K_DIM + 255) / 256, 256, 0, stream>>>(
      W_ih, W_hh, b_ih, b_hh, Wc, bias);
  init_state<<<(B_DIM * H_DIM) / 256, 256, 0, stream>>>(hx0, cx0, x, ab, c_out);

  dim3 grid(G_DIM / 64, B_DIM / 64);          // 16 x 16 = 256 blocks per step
  for (int t = 0; t < STEPS; ++t) {
    lstm_gates<<<grid, 128, 0, stream>>>(ab, Wc, bias, gate);
    // stages x[t+1] (valid: T=512 > STEPS) and writes h for the next GEMM
    lstm_pointwise<<<(B_DIM * H_DIM) / 256, 256, 0, stream>>>(
        gate, c_out, h_out, ab, x + (size_t)(t + 1) * B_DIM * I_DIM);
  }
}
